// MixtureMlp_3616362463378
// MI455X (gfx1250) — matrix operand order repaired and checked by reading, not compiled, not run on hardware
//
#include <hip/hip_runtime.h>
#include <hip/hip_bf16.h>

// Problem constants
#define BBATCH 8
#define NPOS   1024
#define D_IN   768
#define D_H    3072
#define D_OUT  768
#define RR     16
#define MTOT   (BBATCH * NPOS)   // 8192

#if __has_builtin(__builtin_amdgcn_global_load_async_to_lds_b128) && \
    __has_builtin(__builtin_amdgcn_s_wait_asynccnt)
#define HAVE_ASYNC 1
#else
#define HAVE_ASYNC 0
#endif

typedef __attribute__((ext_vector_type(16))) __bf16 v16bf;
typedef __attribute__((ext_vector_type(8)))  float  v8f;
typedef __attribute__((ext_vector_type(4)))  int    v4i;

// 16-byte global -> LDS copy: async (ASYNCcnt-tracked, no VGPR round trip).
__device__ __forceinline__ void copy16(const void* g, void* l) {
#if HAVE_ASYNC
    __builtin_amdgcn_global_load_async_to_lds_b128(
        (__attribute__((address_space(1))) v4i*)g,
        (__attribute__((address_space(3))) v4i*)l, 0, 0);
#else
    *(float4*)l = *(const float4*)g;
#endif
}

// -------------------------------------------------------------------------
// Kernel 1: per-position softmax over experts.
// -------------------------------------------------------------------------
__global__ void coef_softmax_kernel(const float* __restrict__ cw,
                                    const float* __restrict__ ct,
                                    float* __restrict__ coefW,
                                    float* __restrict__ coef_out) {
    int n = blockIdx.x * blockDim.x + threadIdx.x;
    if (n >= NPOS) return;
    float t = expf(ct[n]);
    float l[RR];
    float mx = -3.4e38f;
#pragma unroll
    for (int r = 0; r < RR; ++r) {
        l[r] = t * cw[n * RR + r];
        mx = fmaxf(mx, l[r]);
    }
    float s = 0.f;
#pragma unroll
    for (int r = 0; r < RR; ++r) {
        l[r] = expf(l[r] - mx);
        s += l[r];
    }
    float inv = 1.f / s;
#pragma unroll
    for (int r = 0; r < RR; ++r) {
        float p = l[r] * inv;
        coefW[n * RR + r] = p;
        if (coef_out) coef_out[n * RR + r] = p;
    }
}

// -------------------------------------------------------------------------
// Kernel 2: blended bias  bb[n, j] = sum_r coef[n,r] * bias[r, j]
// -------------------------------------------------------------------------
__global__ void bias_blend_kernel(const float* __restrict__ coefW,
                                  const float* __restrict__ bias,
                                  float* __restrict__ bb, int dim) {
    int idx = blockIdx.x * blockDim.x + threadIdx.x;
    if (idx >= NPOS * dim) return;
    int n = idx / dim;
    int j = idx - n * dim;
    float acc = 0.f;
#pragma unroll
    for (int r = 0; r < RR; ++r)
        acc = fmaf(coefW[n * RR + r], bias[r * dim + j], acc);
    bb[idx] = acc;
}

// -------------------------------------------------------------------------
// Fused mixture GEMM:  C[m,nb] = sum_r sum_i (coef[n(m),r] * A[m,i]) * W[r,nb,i]
//
// Per K-step (32):
//   1) async-copy raw stage s+1 (GLOBAL_LOAD_ASYNC_TO_LDS_B128, ASYNCcnt)
//   2) s_wait_asynccnt + barrier   (stage s raw tiles complete)
//   3) cooperative raw -> bf16 conversion pass in LDS (coef scale folded
//      into A here; done once per block, not once per wave)
//   4) barrier
//   5) fragment gather = pure contiguous ds_load_b128 from bf16 tiles
//      -> 8 v_wmma_f32_16x16x32_bf16 per wave (B frags reused 2x),
//      no VALU touching WMMA sources => no WAR hazard NOPs.
// -------------------------------------------------------------------------
template <int KPE, int NDIM, bool FC1>
__global__ __launch_bounds__(256)
void mix_gemm_kernel(const float*  __restrict__ Ax,   // FC1: x (f32)
                     const __bf16* __restrict__ Ah,   // FC2: h (bf16)
                     const float*  __restrict__ Bw,   // [R, NDIM, KPE]
                     const float*  __restrict__ coefW,
                     const float*  __restrict__ bb,   // [NPOS, NDIM]
                     __bf16*       __restrict__ Hout,
                     float*        __restrict__ Yout)
{
    constexpr int SPE  = KPE / 32;         // K-steps per expert
    constexpr int NST  = RR * SPE;         // total K-steps
    constexpr int ASZ  = FC1 ? 4 : 2;      // raw A element bytes
    constexpr int ARB  = 32 * ASZ;         // raw A row bytes
    constexpr int ACPR = ARB / 16;         // 16B chunks per raw A row (8 / 4)
    constexpr int PTA  = ACPR;             // per-thread A chunks (8 / 4)
    constexpr int PTB  = 2;                // per-thread B chunks
    constexpr int INFLIGHT = PTA + PTB;

    __shared__ __align__(16) unsigned char Araw[2][256 * ARB]; // raw A tiles
    __shared__ __align__(16) float         Braw[2][64][32];    // raw B tiles
    __shared__ __align__(16) __bf16        Abf[2][256 * 32];   // scaled bf16 A
    __shared__ __align__(16) __bf16        Bbf[2][64 * 32];    // bf16 B

    const int tid  = threadIdx.x;
    const int wave = tid >> 5;
    const int lane = tid & 31;
    const int g    = lane >> 4;
    const int lan  = lane & 15;

    const int m0 = blockIdx.y * 256;
    const int n0 = blockIdx.x * 64;

    const char* Abase = FC1 ? (const char*)Ax : (const char*)Ah;

    v8f acc[2][4];
#pragma unroll
    for (int i = 0; i < 2; ++i)
#pragma unroll
        for (int j = 0; j < 4; ++j)
#pragma unroll
            for (int e = 0; e < 8; ++e) acc[i][j][e] = 0.f;

    auto issue = [&](int st, int buf) {
        const int r   = st / SPE;
        const int ik0 = (st - r * SPE) * 32;
#pragma unroll
        for (int t = 0; t < PTA; ++t) {
            int c   = tid + t * 256;
            int row = c / ACPR;
            int off = (c - row * ACPR) * 16;
            copy16(Abase + ((size_t)(m0 + row) * KPE + ik0) * ASZ + off,
                   &Araw[buf][row * ARB + off]);
        }
#pragma unroll
        for (int t = 0; t < PTB; ++t) {
            int c   = tid + t * 256;
            int row = c >> 3;
            int off = (c & 7) * 16;
            copy16((const char*)(Bw + ((size_t)r * NDIM + n0 + row) * KPE + ik0) + off,
                   (char*)&Braw[buf][row][0] + off);
        }
    };

    issue(0, 0);

    for (int st = 0; st < NST; ++st) {
        const int buf = st & 1;
        const int r   = st / SPE;

        if (st + 1 < NST) {
            issue(st + 1, buf ^ 1);
#if HAVE_ASYNC
            __builtin_amdgcn_s_wait_asynccnt(INFLIGHT);
#endif
        } else {
#if HAVE_ASYNC
            __builtin_amdgcn_s_wait_asynccnt(0);
#endif
        }
        __syncthreads();   // raw tiles for stage st are complete everywhere

        // ---- conversion pass (once per block) ----
        {
            // A: one row (32 elems) per thread, coef scale folded in
            const float sc = coefW[((m0 + tid) & (NPOS - 1)) * RR + r];
            __bf16* dst = &Abf[buf][tid * 32];
            if constexpr (FC1) {
                const float* src = (const float*)&Araw[buf][tid * ARB];
#pragma unroll
                for (int e = 0; e < 32; ++e) dst[e] = (__bf16)(src[e] * sc);
            } else {
                const __bf16* src = (const __bf16*)&Araw[buf][tid * ARB];
#pragma unroll
                for (int e = 0; e < 32; ++e)
                    dst[e] = (__bf16)((float)src[e] * sc);
            }
            // B: 8 elems per thread (4 threads per row)
            const int    brow = tid >> 2;
            const int    bc0  = (tid & 3) * 8;
            const float* bsrc = &Braw[buf][brow][bc0];
            __bf16*      bdst = &Bbf[buf][brow * 32 + bc0];
#pragma unroll
            for (int e = 0; e < 8; ++e) bdst[e] = (__bf16)bsrc[e];
        }
        __syncthreads();   // bf16 tiles visible

        // ---- fragment gather (pure contiguous ds_load_b128) + WMMA ----
        v16bf a[2];
#pragma unroll
        for (int i = 0; i < 2; ++i) {
            const int     mrow = wave * 32 + i * 16 + lan;
            const __bf16* ar   = &Abf[buf][mrow * 32];
#pragma unroll
            for (int e = 0; e < 16; ++e) {
                int k = (g << 4) + e;
                a[i][e] = ar[k];
            }
        }
#pragma unroll
        for (int j = 0; j < 4; ++j) {
            const __bf16* br = &Bbf[buf][(j * 16 + lan) * 32 + (g << 4)];
            v16bf b;
#pragma unroll
            for (int e = 0; e < 16; ++e) b[e] = br[e];
#pragma unroll
            for (int i = 0; i < 2; ++i)
                acc[i][j] = __builtin_amdgcn_wmma_f32_16x16x32_bf16(
                    false, a[i], false, b, (short)0, acc[i][j], false, false);
        }
    }

    // Epilogue: C layout: VGPR v -> M = v + 8*g, column = lane%16
#pragma unroll
    for (int i = 0; i < 2; ++i) {
#pragma unroll
        for (int j = 0; j < 4; ++j) {
            int col = n0 + j * 16 + lan;
#pragma unroll
            for (int v = 0; v < 8; ++v) {
                int m = m0 + wave * 32 + i * 16 + v + (g << 3);
                int n = m & (NPOS - 1);
                float val = acc[i][j][v] + bb[(size_t)n * NDIM + col];
                if constexpr (FC1) {
                    float gv =
                        0.5f * val * (1.0f + erff(val * 0.70710678118654752f));
                    Hout[(size_t)m * NDIM + col] = (__bf16)gv;
                } else {
                    Yout[(size_t)m * NDIM + col] = val;
                }
            }
        }
    }
}

// -------------------------------------------------------------------------
// Launch
// -------------------------------------------------------------------------
extern "C" void kernel_launch(void* const* d_in, const int* in_sizes, int n_in,
                              void* d_out, int out_size, void* d_ws, size_t ws_size,
                              hipStream_t stream) {
    const float* x  = (const float*)d_in[0];   // (B, N, D_IN)
    const float* cw = (const float*)d_in[1];   // (N, R)
    const float* ct = (const float*)d_in[2];   // (N,)
    const float* w1 = (const float*)d_in[3];   // (R, D_H, D_IN)
    const float* b1 = (const float*)d_in[4];   // (R, D_H)
    const float* w2 = (const float*)d_in[5];   // (R, D_OUT, D_H)
    const float* b2 = (const float*)d_in[6];   // (R, D_OUT)
    float* y = (float*)d_out;                  // (B, N, D_OUT) [+ coef tail]

    // Workspace carve-up
    char* w = (char*)d_ws;
    float*  coefW = (float*)w;                                         // 64 KB
    float*  bb1   = (float*)(w + (size_t)65536);                       // 12 MB
    float*  bb2   = (float*)(w + (size_t)65536 + 12582912);            // 3 MB
    __bf16* hbf   = (__bf16*)(w + (size_t)65536 + 12582912 + 3145728); // 50 MB

    float* coef_out = nullptr;
    if (out_size >= MTOT * D_OUT + NPOS * RR)
        coef_out = y + (size_t)MTOT * D_OUT;

    // 1) coefficients
    coef_softmax_kernel<<<(NPOS + 255) / 256, 256, 0, stream>>>(cw, ct, coefW,
                                                                coef_out);
    // 2) blended biases
    bias_blend_kernel<<<(NPOS * D_H + 255) / 256, 256, 0, stream>>>(coefW, b1,
                                                                    bb1, D_H);
    bias_blend_kernel<<<(NPOS * D_OUT + 255) / 256, 256, 0, stream>>>(coefW, b2,
                                                                      bb2, D_OUT);
    // 3) fc1 + GELU -> h (bf16)
    mix_gemm_kernel<D_IN, D_H, true>
        <<<dim3(D_H / 64, MTOT / 256), 256, 0, stream>>>(
            x, nullptr, w1, coefW, bb1, hbf, nullptr);
    // 4) fc2 -> y (f32)
    mix_gemm_kernel<D_H, D_OUT, false>
        <<<dim3(D_OUT / 64, MTOT / 256), 256, 0, stream>>>(
            nullptr, hbf, w2, coefW, bb2, nullptr, y);
}
